// ContrastiveLoss_62689342652556
// MI455X (gfx1250) — matrix operand order repaired and checked by reading, not compiled, not run on hardware
//
#include <hip/hip_runtime.h>

// ---------------- problem constants ----------------
#define BATCH 8192
#define DIM   256
#define INV_T 1000.0f           // 1/TEMP
#define NSPLIT 4                // j-range splits for parallelism
#define JCHUNK (BATCH / NSPLIT) // 2048
#define NTILES (JCHUNK / 16)    // 128 j-tiles per chunk
#define WAVES_PER_BLOCK 8
#define BN_WAVE 16
#define BN_BLOCK (WAVES_PER_BLOCK * BN_WAVE) // 128 i-rows per block

typedef __attribute__((ext_vector_type(16))) __bf16 v16bf;
typedef __attribute__((ext_vector_type(8)))  __bf16 v8bf;
typedef __attribute__((ext_vector_type(8)))  float  v8f;

// CDNA5 async global->LDS path (ASYNCcnt), guarded so the file compiles on any toolchain
#if defined(__has_builtin)
#if __has_builtin(__builtin_amdgcn_global_load_async_to_lds_b128) && \
    __has_builtin(__builtin_amdgcn_s_wait_asynccnt)
#define USE_ASYNC 1
#endif
#endif
#ifndef USE_ASYNC
#define USE_ASYNC 0
#endif

#if USE_ASYNC
// exact parameter types per hipcc diagnostic:
//   arg0: 'int __attribute__((__vector_size__(4*sizeof(int)))) __device__ *'  (AS1 = global)
//   arg1: LDS pointer (AS3)
typedef int v4i_raw __attribute__((__vector_size__(4 * sizeof(int))));
typedef __attribute__((address_space(1))) v4i_raw as1_v4i;
typedef __attribute__((address_space(3))) v4i_raw as3_v4i;
#endif

// float -> bf16 (round-to-nearest-even), bf16 -> float, via bit ops only
__device__ __forceinline__ unsigned short f2bf(float x) {
  unsigned u = __float_as_uint(x);
  u += 0x7FFFu + ((u >> 16) & 1u);
  return (unsigned short)(u >> 16);
}
__device__ __forceinline__ float bf2f(unsigned short h) {
  return __uint_as_float(((unsigned)h) << 16);
}

// ---------------- phase 1: normalize rows, split fp32 -> bf16 hi/lo, diag dot ----------------
__global__ void __launch_bounds__(256)
simclr_normalize_kernel(const float* __restrict__ p1, const float* __restrict__ p2,
                        unsigned short* __restrict__ zi_hi, unsigned short* __restrict__ zi_lo,
                        unsigned short* __restrict__ zj_hi, unsigned short* __restrict__ zj_lo,
                        float* __restrict__ pos) {
  const int wave = threadIdx.x >> 5;
  const int lane = threadIdx.x & 31;
  const int row  = blockIdx.x * WAVES_PER_BLOCK + wave;
  const float* r1 = p1 + row * DIM;
  const float* r2 = p2 + row * DIM;

  float a[8], b[8];
  float s1 = 0.f, s2 = 0.f, dt = 0.f;
#pragma unroll
  for (int t = 0; t < 8; ++t) {
    const int k = t * 32 + lane;
    a[t] = r1[k]; b[t] = r2[k];
    s1 = fmaf(a[t], a[t], s1);
    s2 = fmaf(b[t], b[t], s2);
    dt = fmaf(a[t], b[t], dt);
  }
#pragma unroll
  for (int m = 16; m >= 1; m >>= 1) {   // wave32 reduction
    s1 += __shfl_xor(s1, m, 32);
    s2 += __shfl_xor(s2, m, 32);
    dt += __shfl_xor(dt, m, 32);
  }
  const float inv1 = 1.0f / fmaxf(sqrtf(s1), 1e-12f);
  const float inv2 = 1.0f / fmaxf(sqrtf(s2), 1e-12f);
  if (lane == 0) pos[row] = dt * inv1 * inv2;

#pragma unroll
  for (int t = 0; t < 8; ++t) {
    const int k   = t * 32 + lane;
    const int idx = row * DIM + k;
    float x = a[t] * inv1;
    unsigned short h = f2bf(x);
    zi_hi[idx] = h;
    zi_lo[idx] = f2bf(x - bf2f(h));
    float y = b[t] * inv2;
    unsigned short h2 = f2bf(y);
    zj_hi[idx] = h2;
    zj_lo[idx] = f2bf(y - bf2f(h2));
  }
}

// ---------------- fragment loaders (contiguous k order, same for A and B) ----------------
// A 16x32 bf16 (MxK): lane group g = lane>>4, M = lane&15.
//   elements 0..7  -> K = k0 + g*16 + e        (16 contiguous bytes)
//   elements 8..15 -> K = k0 + g*16 + e        (16 contiguous bytes)
__device__ __forceinline__ v16bf lds_frag_A(const unsigned short* t, int nl, int k0, int g) {
  const __bf16* base = (const __bf16*)(t + nl * DIM + k0 + g * 16);
  v8bf lo = *(const v8bf*)(base);          // ds_load_b128
  v8bf hi = *(const v8bf*)(base + 8);      // ds_load_b128
  v16bf r;
#pragma unroll
  for (int e = 0; e < 8; ++e) { r[e] = lo[e]; r[e + 8] = hi[e]; }
  return r;
}
// B 32x16 bf16 (KxN): N = lane&15; lanes 0-15 hold K=k0..k0+15, lanes 16-31 K=k0+16..k0+31.
__device__ __forceinline__ v16bf load_frag_B(const unsigned short* z, int row, int k0, int g) {
  return *(const v16bf*)(z + row * DIM + k0 + g * 16);
}

// Cooperative copy of one contiguous 16-row j-tile (hi+lo, 8KB each) into LDS.
// 256 threads x 4 x 16B. Async (ASYNCcnt) when available, sync load+ds_store otherwise.
__device__ __forceinline__ void stage_tile(const unsigned short* gh, const unsigned short* gl,
                                           unsigned short* lh, unsigned short* ll, int tid) {
#if USE_ASYNC
  __builtin_amdgcn_global_load_async_to_lds_b128(
      (as1_v4i*)(gh + tid * 8),         (as3_v4i*)(lh + tid * 8), 0, 0);
  __builtin_amdgcn_global_load_async_to_lds_b128(
      (as1_v4i*)(gh + (tid + 256) * 8), (as3_v4i*)(lh + (tid + 256) * 8), 0, 0);
  __builtin_amdgcn_global_load_async_to_lds_b128(
      (as1_v4i*)(gl + tid * 8),         (as3_v4i*)(ll + tid * 8), 0, 0);
  __builtin_amdgcn_global_load_async_to_lds_b128(
      (as1_v4i*)(gl + (tid + 256) * 8), (as3_v4i*)(ll + (tid + 256) * 8), 0, 0);
#else
  *(v8bf*)(lh + tid * 8)         = *(const v8bf*)(gh + tid * 8);
  *(v8bf*)(lh + (tid + 256) * 8) = *(const v8bf*)(gh + (tid + 256) * 8);
  *(v8bf*)(ll + tid * 8)         = *(const v8bf*)(gl + tid * 8);
  *(v8bf*)(ll + (tid + 256) * 8) = *(const v8bf*)(gl + (tid + 256) * 8);
#endif
}

// ---------------- phase 2: fused GEMM + online logsumexp over j ----------------
// C[m][n] = sum_k zj[j0+m][k] * zi[i0+n][k]  ==> sim[i0+n][j0+m]
// A (z_j tile) staged through double-buffered LDS, shared by all 8 waves.
// B (z_i, this wave's 16 i-rows, full K, hi+lo) resident in 128 VGPRs.
__global__ void __launch_bounds__(256)
simclr_gemm_lse_kernel(const unsigned short* __restrict__ zi_hi, const unsigned short* __restrict__ zi_lo,
                       const unsigned short* __restrict__ zj_hi, const unsigned short* __restrict__ zj_lo,
                       float* __restrict__ part_m, float* __restrict__ part_s) {
  __shared__ __align__(16) unsigned short lds_h[2][16 * DIM];  // 2 x 8KB
  __shared__ __align__(16) unsigned short lds_l[2][16 * DIM];  // 2 x 8KB

  const int tid   = threadIdx.x;
  const int wave  = tid >> 5;
  const int lane  = tid & 31;
  const int g     = lane >> 4;
  const int nl    = lane & 15;
  const int i0    = blockIdx.x * BN_BLOCK + wave * BN_WAVE;
  const int jbase = blockIdx.y * JCHUNK;

  // Preload this wave's B fragments for the full K=256 (hi+lo): 8 chunks x 2 x 8 VGPRs
  v16bf bhi[8], blo[8];
#pragma unroll
  for (int c8 = 0; c8 < 8; ++c8) {
    bhi[c8] = load_frag_B(zi_hi, i0 + nl, c8 * 32, g);
    blo[c8] = load_frag_B(zi_lo, i0 + nl, c8 * 32, g);
  }

  // stage tile 0
  stage_tile(zj_hi + (size_t)jbase * DIM, zj_lo + (size_t)jbase * DIM,
             &lds_h[0][0], &lds_l[0][0], tid);

  float run_m = -__builtin_inff();
  float run_s = 0.0f;

  for (int jt = 0; jt < NTILES; ++jt) {
    const int buf = jt & 1;
    if (jt + 1 < NTILES) {  // prefetch next tile into the other buffer, then wait on current
      const size_t goff = (size_t)(jbase + (jt + 1) * 16) * DIM;
      stage_tile(zj_hi + goff, zj_lo + goff, &lds_h[buf ^ 1][0], &lds_l[buf ^ 1][0], tid);
#if USE_ASYNC
      __builtin_amdgcn_s_wait_asynccnt(4);   // 4 just-issued may stay in flight
#endif
    } else {
#if USE_ASYNC
      __builtin_amdgcn_s_wait_asynccnt(0);
#endif
    }
    __syncthreads();  // tile jt fully in LDS for all waves

    const unsigned short* th = &lds_h[buf][0];
    const unsigned short* tl = &lds_l[buf][0];

    v8f c = {};
    // register-pipelined chunk loop: ds_loads for chunk c8+1 issue before WMMAs of chunk c8
    v16bf ahi_c = lds_frag_A(th, nl, 0, g);
    v16bf alo_c = lds_frag_A(tl, nl, 0, g);
#pragma unroll
    for (int c8 = 0; c8 < 8; ++c8) {
      v16bf ahi_n = ahi_c, alo_n = alo_c;
      if (c8 < 7) {
        ahi_n = lds_frag_A(th, nl, (c8 + 1) * 32, g);
        alo_n = lds_frag_A(tl, nl, (c8 + 1) * 32, g);
      }
      // bf16x3: hi*hi + hi*lo + lo*hi  (~fp32-accurate dot)
      c = __builtin_amdgcn_wmma_f32_16x16x32_bf16(false, ahi_c, false, bhi[c8], (short)0, c, false, false);
      c = __builtin_amdgcn_wmma_f32_16x16x32_bf16(false, ahi_c, false, blo[c8], (short)0, c, false, false);
      c = __builtin_amdgcn_wmma_f32_16x16x32_bf16(false, alo_c, false, bhi[c8], (short)0, c, false, false);
      ahi_c = ahi_n;
      alo_c = alo_n;
    }

    // online logsumexp update over the 8 j's this lane owns
    float tm = c[0];
#pragma unroll
    for (int r = 1; r < 8; ++r) tm = fmaxf(tm, c[r]);
    const float nm = fmaxf(run_m, tm);
    run_s *= __expf((run_m - nm) * INV_T);
#pragma unroll
    for (int r = 0; r < 8; ++r) run_s += __expf((c[r] - nm) * INV_T);
    run_m = nm;

    __syncthreads();  // all waves done reading buf before it is overwritten next iter
  }

  // combine the two half-wave lanes holding the same i (rows m and m+8)
  const float om = __shfl_xor(run_m, 16, 32);
  const float os = __shfl_xor(run_s, 16, 32);
  const float M  = fmaxf(run_m, om);
  const float S  = run_s * __expf((run_m - M) * INV_T) + os * __expf((om - M) * INV_T);

  if (lane < 16) {
    const int i = i0 + nl;
    part_m[i * NSPLIT + blockIdx.y] = M;
    part_s[i * NSPLIT + blockIdx.y] = S;
  }
}

// ---------------- phase 3: merge j-splits, per-row loss ----------------
__global__ void __launch_bounds__(256)
simclr_finalize_kernel(const float* __restrict__ part_m, const float* __restrict__ part_s,
                       const float* __restrict__ pos, float* __restrict__ loss) {
  const int i = blockIdx.x * blockDim.x + threadIdx.x;
  if (i >= BATCH) return;
  float M = part_m[i * NSPLIT];
#pragma unroll
  for (int c = 1; c < NSPLIT; ++c) M = fmaxf(M, part_m[i * NSPLIT + c]);
  float S = 0.f;
#pragma unroll
  for (int c = 0; c < NSPLIT; ++c)
    S += part_s[i * NSPLIT + c] * __expf((part_m[i * NSPLIT + c] - M) * INV_T);
  // loss_i = -pos_i/T + lse_i = (M - pos_i)/T + log(S)
  loss[i] = (M - pos[i]) * INV_T + logf(S);
}

// ---------------- phase 4: deterministic scalar reduction ----------------
__global__ void __launch_bounds__(256)
simclr_reduce_kernel(const float* __restrict__ loss, const float* __restrict__ pos,
                     float* __restrict__ out) {
  __shared__ float sl[256], sp[256];
  const int t = threadIdx.x;
  float l = 0.f, p = 0.f;
  for (int k = t; k < BATCH; k += 256) { l += loss[k]; p += pos[k]; }
  sl[t] = l; sp[t] = p;
  __syncthreads();
  for (int s = 128; s >= 1; s >>= 1) {
    if (t < s) { sl[t] += sl[t + s]; sp[t] += sp[t + s]; }
    __syncthreads();
  }
  if (t == 0) { out[0] = sl[0] / (float)BATCH; out[1] = sp[0]; }
}

// ---------------- host launcher ----------------
extern "C" void kernel_launch(void* const* d_in, const int* in_sizes, int n_in,
                              void* d_out, int out_size, void* d_ws, size_t ws_size,
                              hipStream_t stream) {
  const float* p1 = (const float*)d_in[0];
  const float* p2 = (const float*)d_in[1];
  float* out = (float*)d_out;

  // workspace layout (bytes): 4x 4MB bf16 operand copies + small per-row arrays
  char* ws = (char*)d_ws;
  const size_t BF = (size_t)BATCH * DIM * sizeof(unsigned short); // 4 MB each
  unsigned short* zi_hi = (unsigned short*)(ws);
  unsigned short* zi_lo = (unsigned short*)(ws + 1 * BF);
  unsigned short* zj_hi = (unsigned short*)(ws + 2 * BF);
  unsigned short* zj_lo = (unsigned short*)(ws + 3 * BF);
  char* tail = ws + 4 * BF;
  float* pos    = (float*)(tail);                                  // BATCH
  float* part_m = (float*)(tail + BATCH * sizeof(float));          // BATCH*NSPLIT
  float* part_s = (float*)(tail + BATCH * (1 + NSPLIT) * sizeof(float));
  float* loss   = (float*)(tail + BATCH * (1 + 2 * NSPLIT) * sizeof(float));

  simclr_normalize_kernel<<<BATCH / WAVES_PER_BLOCK, 256, 0, stream>>>(
      p1, p2, zi_hi, zi_lo, zj_hi, zj_lo, pos);

  dim3 grid2(BATCH / BN_BLOCK, NSPLIT);
  simclr_gemm_lse_kernel<<<grid2, 256, 0, stream>>>(
      zi_hi, zi_lo, zj_hi, zj_lo, part_m, part_s);

  simclr_finalize_kernel<<<BATCH / 256, 256, 0, stream>>>(part_m, part_s, pos, loss);

  simclr_reduce_kernel<<<1, 256, 0, stream>>>(loss, pos, out);
}
